// CCT_37125697306958
// MI455X (gfx1250) — compile-verified, matrix operand order checked by reading, not run on hardware
//
#include <hip/hip_runtime.h>
#include <hip/hip_bf16.h>
#include <cstdint>
#include <cmath>

// ---------------------------------------------------------------------------
// CCT cross-attention block for MI455X (gfx1250).
// All GEMMs run through v_wmma_f32_16x16x32_bf16 (wave32 WMMA), fp32 accum.
// Roofline: ~235 GFLOP vs ~0.5 GB traffic -> matrix-core bound; bf16 keeps
// f32 accumulation fidelity on the XDL WMMA path.
// GEMM staging uses GLOBAL_LOAD_ASYNC_TO_LDS_B128 (ASYNCcnt) with
// double-buffered LDS: one barrier per K-step, no VGPR round trip.
// ---------------------------------------------------------------------------

typedef unsigned short u16;
typedef unsigned int   u32;
typedef long long      s64;

typedef __attribute__((ext_vector_type(16))) __bf16 v16bf;
typedef __attribute__((ext_vector_type(8)))  float  v8f;
typedef __attribute__((ext_vector_type(4)))  int    v4i;

union FragBF { u32 u[8]; v16bf v; };

#define AS_G __attribute__((address_space(1)))
#define AS_L __attribute__((address_space(3)))

#if __has_builtin(__builtin_amdgcn_global_load_async_to_lds_b128)
#define HAVE_ASYNC_LDS 1
#else
#define HAVE_ASYNC_LDS 0
#endif

__device__ __forceinline__ u16 f2bf(float f) {
  u32 u = __float_as_uint(f);
  u32 r = u + 0x7FFFu + ((u >> 16) & 1u);   // round-to-nearest-even
  return (u16)(r >> 16);
}
__device__ __forceinline__ float bf2f(u16 s) {
  return __uint_as_float(((u32)s) << 16);
}
__device__ __forceinline__ float gelu_exact(float x) {
  return 0.5f * x * (1.0f + erff(x * 0.70710678118654752440f));
}

// -------------------------------- convert ----------------------------------
__global__ __launch_bounds__(256) void convert_f32_bf16_kernel(
    const float* __restrict__ src, u16* __restrict__ dst, s64 n) {
  s64 i = (s64)blockIdx.x * 1024 + (s64)threadIdx.x * 4;
#pragma unroll
  for (int t = 0; t < 4; ++t) {
    s64 j = i + t;
    if (j < n) dst[j] = f2bf(src[j]);
  }
}

// ------------------------------- layernorm ---------------------------------
__global__ __launch_bounds__(256) void ln_rows_kernel(
    const float* __restrict__ x, const float* __restrict__ g,
    const float* __restrict__ b, u16* __restrict__ out, int D, float eps) {
  const int row = blockIdx.x;
  const int tid = threadIdx.x;
  const float* xr = x + (s64)row * D;
  __shared__ float rs[256], rq[256];
  float s = 0.f, q = 0.f;
  for (int j = tid; j < D; j += 256) { float v = xr[j]; s += v; q += v * v; }
  rs[tid] = s; rq[tid] = q; __syncthreads();
  for (int st = 128; st > 0; st >>= 1) {
    if (tid < st) { rs[tid] += rs[tid + st]; rq[tid] += rq[tid + st]; }
    __syncthreads();
  }
  float mean = rs[0] / (float)D;
  float var  = rq[0] / (float)D - mean * mean;
  float rstd = rsqrtf(var + eps);
  u16* po = out + (s64)row * D;
  for (int j = tid; j < D; j += 256)
    po[j] = f2bf((xr[j] - mean) * rstd * g[j] + b[j]);
}

// concat(x1..x4) + layernorm over 960, output bf16
__global__ __launch_bounds__(256) void ln_kv_kernel(
    const float* __restrict__ x1, const float* __restrict__ x2,
    const float* __restrict__ x3, const float* __restrict__ x4,
    const float* __restrict__ g, const float* __restrict__ b,
    u16* __restrict__ out, float eps) {
  const int row = blockIdx.x;
  const int tid = threadIdx.x;
  __shared__ float rs[256], rq[256];
  float vloc[4];
  int cnt = 0;
  float s = 0.f, q = 0.f;
  for (int j = tid; j < 960; j += 256) {
    float v;
    if (j < 64)       v = x1[(s64)row * 64  + j];
    else if (j < 192) v = x2[(s64)row * 128 + (j - 64)];
    else if (j < 448) v = x3[(s64)row * 256 + (j - 192)];
    else              v = x4[(s64)row * 512 + (j - 448)];
    vloc[cnt++] = v; s += v; q += v * v;
  }
  rs[tid] = s; rq[tid] = q; __syncthreads();
  for (int st = 128; st > 0; st >>= 1) {
    if (tid < st) { rs[tid] += rs[tid + st]; rq[tid] += rq[tid + st]; }
    __syncthreads();
  }
  float mean = rs[0] / 960.f;
  float var  = rq[0] / 960.f - mean * mean;
  float rstd = rsqrtf(var + eps);
  u16* po = out + (s64)row * 960;
  cnt = 0;
  for (int j = tid; j < 960; j += 256)
    po[j] = f2bf((vloc[cnt++] - mean) * rstd * g[j] + b[j]);
}

// ------------------- instance-norm stats + softmax -------------------------
__global__ __launch_bounds__(256) void inorm_stats_kernel(
    const float* __restrict__ a, float* __restrict__ stats, s64 len) {
  const int z = blockIdx.x;
  const int tid = threadIdx.x;
  const float* p = a + (s64)z * len;
  __shared__ float rs[256], rq[256];
  float s = 0.f, q = 0.f;
  for (s64 j = tid; j < len; j += 256) { float v = p[j]; s += v; q += v * v; }
  rs[tid] = s; rq[tid] = q; __syncthreads();
  for (int st = 128; st > 0; st >>= 1) {
    if (tid < st) { rs[tid] += rs[tid + st]; rq[tid] += rq[tid + st]; }
    __syncthreads();
  }
  if (tid == 0) {
    float m = rs[0] / (float)len;
    float var = rq[0] / (float)len - m * m;
    stats[2 * z]     = m;
    stats[2 * z + 1] = rsqrtf(var + 1e-5f);
  }
}

__global__ __launch_bounds__(256) void inorm_softmax_kernel(
    const float* __restrict__ a, const float* __restrict__ stats,
    u16* __restrict__ out, int cdim) {
  const int r = blockIdx.x;          // 0 .. 32*cdim-1
  const int z = r / cdim;
  const int tid = threadIdx.x;
  const float* pr = a + (s64)r * 960;
  const float mean = stats[2 * z];
  const float rstd = stats[2 * z + 1];
  float vloc[4];
  int cnt = 0;
  float mx = -3.0e38f;
  for (int j = tid; j < 960; j += 256) {
    float v = (pr[j] - mean) * rstd;
    vloc[cnt++] = v;
    mx = fmaxf(mx, v);
  }
  __shared__ float red[256];
  red[tid] = mx; __syncthreads();
  for (int st = 128; st > 0; st >>= 1) {
    if (tid < st) red[tid] = fmaxf(red[tid], red[tid + st]);
    __syncthreads();
  }
  mx = red[0]; __syncthreads();
  float s = 0.f;
  for (int t = 0; t < cnt; ++t) { float e = __expf(vloc[t] - mx); vloc[t] = e; s += e; }
  red[tid] = s; __syncthreads();
  for (int st = 128; st > 0; st >>= 1) {
    if (tid < st) red[tid] += red[tid + st];
    __syncthreads();
  }
  float inv = 1.f / red[0];
  u16* po = out + (s64)r * 960;
  cnt = 0;
  for (int j = tid; j < 960; j += 256) po[j] = f2bf(vloc[cnt++] * inv);
}

// ------------------------------- transpose ---------------------------------
// batched [Z][R][Cc] -> [Z][Cc][R], bf16
__global__ __launch_bounds__(256) void transpose_bf16_kernel(
    const u16* __restrict__ in, u16* __restrict__ out, int R, int Cc) {
  __shared__ u16 t[32][33];
  const s64 zo = (s64)blockIdx.z * R * Cc;
  const int c0 = blockIdx.x * 32, r0 = blockIdx.y * 32;
  const int lx = threadIdx.x & 31, ly = threadIdx.x >> 5;
#pragma unroll
  for (int i = 0; i < 32; i += 8) {
    int r = r0 + ly + i, c = c0 + lx;
    if (r < R && c < Cc) t[ly + i][lx] = in[zo + (s64)r * Cc + c];
  }
  __syncthreads();
#pragma unroll
  for (int i = 0; i < 32; i += 8) {
    int r = c0 + ly + i, c = r0 + lx;
    if (r < Cc && c < R) out[zo + (s64)r * R + c] = t[lx][ly + i];
  }
}

// -------------------------------- WMMA GEMM --------------------------------
// C[z][m][n] = scale * sum_k A[z][m][k] * B[z%bmod][n][k]  (+bias/gelu/resid)
// A,B bf16 row-major with leading dim K. 128x128 block, 8 waves of 64x32,
// v_wmma_f32_16x16x32_bf16. Double-buffered LDS (stride 40 -> 80B rows:
// 16B-aligned, conflict-free). Staging via global_load_async_to_lds_b128
// when available (ASYNCcnt), else uint4+ds_store fallback. Out-of-range rows
// are clamped (not predicated): they only feed accumulator rows/cols that the
// store guards drop, so the inner loop has no exec-mask branches.
enum {
  GF_BIAS = 1, GF_GELU = 2, GF_RESF = 4, GF_RESB = 8, GF_OUTF = 16, GF_OUTB = 32
};

__global__ __launch_bounds__(256) void gemm_bf16_wmma_kernel(
    const u16* __restrict__ A, const u16* __restrict__ Bm,
    float* __restrict__ C32, u16* __restrict__ C16,
    const float* __restrict__ bias, const float* __restrict__ resf,
    const u16* __restrict__ resb,
    int M, int N, int K, s64 sA, s64 sB, int bmod, s64 sC, s64 sR,
    float scale, int flags) {
  __shared__ u16 lsA[2][128 * 40];
  __shared__ u16 lsB[2][128 * 40];
  const int z = blockIdx.z;
  const u16* Ab = A + (s64)z * sA;
  const u16* Bb = Bm + (s64)(z % bmod) * sB;
  const int bm = blockIdx.y * 128, bn = blockIdx.x * 128;
  const int tid = threadIdx.x;
  const int lane = tid & 31, wave = tid >> 5;
  const int ll = lane & 15, lh = lane >> 4;
  const int wm = (wave >> 2) * 64, wn = (wave & 3) * 32;
  const int arow = tid >> 1, acol = (tid & 1) * 16;

  // Clamped staging rows: OOB rows load valid-but-unused data.
  int gmc = bm + arow; if (gmc > M - 1) gmc = M - 1;
  int gnc = bn + arow; if (gnc > N - 1) gnc = N - 1;
  const u16* gA = Ab + (s64)gmc * K + acol;
  const u16* gB = Bb + (s64)gnc * K + acol;
  const int lofs = arow * 40 + acol;

  v8f acc[4][2];
#pragma unroll
  for (int tm = 0; tm < 4; ++tm)
#pragma unroll
    for (int tn = 0; tn < 2; ++tn)
#pragma unroll
      for (int v = 0; v < 8; ++v) acc[tm][tn][v] = 0.f;

  auto issue_stage = [&](int k0, int buf) {
#if HAVE_ASYNC_LDS
    const u16* ga = gA + k0;
    const u16* gb = gB + k0;
    u16* la = &lsA[buf][lofs];
    u16* lb = &lsB[buf][lofs];
    __builtin_amdgcn_global_load_async_to_lds_b128(
        (AS_G v4i*)ga, (AS_L v4i*)la, 0, 0);
    __builtin_amdgcn_global_load_async_to_lds_b128(
        (AS_G v4i*)(ga + 8), (AS_L v4i*)(la + 8), 0, 0);
    __builtin_amdgcn_global_load_async_to_lds_b128(
        (AS_G v4i*)gb, (AS_L v4i*)lb, 0, 0);
    __builtin_amdgcn_global_load_async_to_lds_b128(
        (AS_G v4i*)(gb + 8), (AS_L v4i*)(lb + 8), 0, 0);
#else
    const uint4* pa = (const uint4*)(gA + k0);
    const uint4* pb = (const uint4*)(gB + k0);
    uint4 a0 = pa[0], a1 = pa[1];
    uint4 b0 = pb[0], b1 = pb[1];
    *(uint4*)&lsA[buf][lofs]     = a0;
    *(uint4*)&lsA[buf][lofs + 8] = a1;
    *(uint4*)&lsB[buf][lofs]     = b0;
    *(uint4*)&lsB[buf][lofs + 8] = b1;
#endif
  };
  auto wait_stage = [&]() {
#if HAVE_ASYNC_LDS
#if __has_builtin(__builtin_amdgcn_s_wait_asynccnt)
    __builtin_amdgcn_s_wait_asynccnt(0);
#else
    asm volatile("s_wait_asynccnt 0x0" ::: "memory");
#endif
#endif
    __syncthreads();
  };

  const int nst = K >> 5;      // K is a multiple of 32 for every call site
  issue_stage(0, 0);
  wait_stage();

  for (int s = 0; s < nst; ++s) {
    const int cur = s & 1;
    const bool more = (s + 1) < nst;
    if (more) issue_stage((s + 1) << 5, cur ^ 1);

    // Fragment gather per documented 16-bit A-matrix layout:
    // lane ll holds row M (or col N for B); K pairs packed 2/VGPR; lane-half
    // swaps K groups (lh=1 -> K+8). Row stride 40 u16 = 20 u32.
    FragBF fa[4], fb[2];
    const u32* a32 = (const u32*)lsA[cur];
    const u32* b32 = (const u32*)lsB[cur];
#pragma unroll
    for (int tm = 0; tm < 4; ++tm) {
      int base = (wm + tm * 16 + ll) * 20 + lh * 4;
#pragma unroll
      for (int j = 0; j < 4; ++j) {
        fa[tm].u[j]     = a32[base + j];
        fa[tm].u[4 + j] = a32[base + 8 + j];
      }
    }
#pragma unroll
    for (int tn = 0; tn < 2; ++tn) {
      int base = (wn + tn * 16 + ll) * 20 + lh * 4;
#pragma unroll
      for (int j = 0; j < 4; ++j) {
        fb[tn].u[j]     = b32[base + j];
        fb[tn].u[4 + j] = b32[base + 8 + j];
      }
    }
#pragma unroll
    for (int tm = 0; tm < 4; ++tm)
#pragma unroll
      for (int tn = 0; tn < 2; ++tn)
        acc[tm][tn] = __builtin_amdgcn_wmma_f32_16x16x32_bf16(
            false, fa[tm].v, false, fb[tn].v, (short)0, acc[tm][tn],
            false, false);

    if (more) wait_stage();
  }

  // Epilogue. C tile layout: VGPR v -> row v (+8 for lanes 16-31), col = ll.
#pragma unroll
  for (int tm = 0; tm < 4; ++tm) {
    int rbase = bm + wm + tm * 16 + 8 * lh;
#pragma unroll
    for (int tn = 0; tn < 2; ++tn) {
      int gn = bn + wn + tn * 16 + ll;
      if (gn < N) {
#pragma unroll
        for (int v = 0; v < 8; ++v) {
          int gm = rbase + v;
          if (gm < M) {
            float xv = acc[tm][tn][v] * scale;
            if (flags & GF_BIAS) xv += bias[gn];
            if (flags & GF_GELU) xv = gelu_exact(xv);
            s64 ri = (s64)z * sR + (s64)gm * N + gn;
            if (flags & GF_RESF) xv += resf[ri];
            if (flags & GF_RESB) xv += bf2f(resb[ri]);
            s64 ci = (s64)z * sC + (s64)gm * N + gn;
            if (flags & GF_OUTF) C32[ci] = xv;
            if (flags & GF_OUTB) C16[ci] = f2bf(xv);
          }
        }
      }
    }
  }
}

// ------------------------------ orchestration ------------------------------
extern "C" void kernel_launch(void* const* d_in, const int* in_sizes, int n_in,
                              void* d_out, int out_size, void* d_ws, size_t ws_size,
                              hipStream_t stream) {
  (void)in_sizes; (void)n_in; (void)out_size; (void)ws_size;
  const int CSv[4] = {64, 128, 256, 512};
  const int T = 16384;   // B*N tokens
  const int Z = 32;      // B*H batches

  // ---- inputs (setup_inputs dict order) ----
  const float* X[4] = {(const float*)d_in[0], (const float*)d_in[1],
                       (const float*)d_in[2], (const float*)d_in[3]};
  const float* Wk = (const float*)d_in[4];
  const float* Wv = (const float*)d_in[5];
  const float *Wq[4], *Wo[4], *ln1g[4], *ln1b[4], *ln2g[4], *ln2b[4];
  const float *f1w[4], *f1b[4], *f2w[4], *f2b[4];
  for (int i = 0; i < 4; ++i) {
    int base = 6 + 10 * i;
    Wq[i]   = (const float*)d_in[base + 0];
    Wo[i]   = (const float*)d_in[base + 1];
    ln1g[i] = (const float*)d_in[base + 2];
    ln1b[i] = (const float*)d_in[base + 3];
    ln2g[i] = (const float*)d_in[base + 4];
    ln2b[i] = (const float*)d_in[base + 5];
    f1w[i]  = (const float*)d_in[base + 6];
    f1b[i]  = (const float*)d_in[base + 7];
    f2w[i]  = (const float*)d_in[base + 8];
    f2b[i]  = (const float*)d_in[base + 9];
  }
  const float* lnkvg = (const float*)d_in[46];
  const float* lnkvb = (const float*)d_in[47];

  // ---- workspace carve-up ----
  char* ws = (char*)d_ws;
  size_t off = 0;
  auto alloc = [&](s64 bytes) -> char* {
    char* p = ws + off;
    off = (off + (size_t)bytes + 255) & ~(size_t)255;
    return p;
  };
  u16* qbf[4];
  for (int i = 0; i < 4; ++i) qbf[i] = (u16*)alloc((s64)T * CSv[i] * 2);
  u16* kvbf  = (u16*)alloc((s64)T * 960 * 2);
  u16* wkbf  = (u16*)alloc((s64)8 * 960 * 960 * 2);
  u16* wvbf  = (u16*)alloc((s64)8 * 960 * 960 * 2);
  u16 *wqbf[4], *wobf[4], *f1wbf[4], *f2wbf[4];
  for (int i = 0; i < 4; ++i) {
    s64 c = CSv[i];
    wqbf[i]  = (u16*)alloc(8 * c * c * 2);
    wobf[i]  = (u16*)alloc(c * c * 2);
    f1wbf[i] = (u16*)alloc(4 * c * c * 2);
    f2wbf[i] = (u16*)alloc(4 * c * c * 2);
  }
  u16* t1bf   = (u16*)alloc((s64)T * 960 * 2);
  u16* kvpbf  = (u16*)alloc((s64)T * 960 * 2);
  u16* ktbf   = (u16*)alloc((s64)T * 960 * 2);
  u16* qpbf   = (u16*)alloc((s64)T * 512 * 2);
  u16* qptbf  = (u16*)alloc((s64)T * 512 * 2);
  float* scoresf = (float*)alloc((s64)Z * 512 * 960 * 4);
  u16* probbf = (u16*)alloc((s64)Z * 512 * 960 * 2);
  u16* obf    = (u16*)alloc((s64)T * 512 * 2);
  float* attnf = (float*)alloc((s64)T * 512 * 4);
  u16* ln2bf  = (u16*)alloc((s64)T * 512 * 2);
  u16* hbf    = (u16*)alloc((s64)T * 2048 * 2);
  float* stats = (float*)alloc(64 * 4);

  auto conv = [&](const float* src, u16* dst, s64 n) {
    int blocks = (int)((n + 1023) / 1024);
    convert_f32_bf16_kernel<<<blocks, 256, 0, stream>>>(src, dst, n);
  };
  auto gemm = [&](const u16* A, const u16* Bp, float* C32, u16* C16,
                  const float* bias, const float* rf, const u16* rb,
                  int M, int Nn, int K, s64 sA, s64 sB, int bmod,
                  s64 sC, s64 sR, int batch, float scale, int flags) {
    dim3 grid((Nn + 127) / 128, (M + 127) / 128, batch);
    gemm_bf16_wmma_kernel<<<grid, 256, 0, stream>>>(
        A, Bp, C32, C16, bias, rf, rb, M, Nn, K, sA, sB, bmod, sC, sR, scale,
        flags);
  };

  // ---- weight conversion ----
  conv(Wk, wkbf, (s64)8 * 960 * 960);
  conv(Wv, wvbf, (s64)8 * 960 * 960);
  for (int i = 0; i < 4; ++i) {
    s64 c = CSv[i];
    conv(Wq[i], wqbf[i], 8 * c * c);
    conv(Wo[i], wobf[i], c * c);
    conv(f1w[i], f1wbf[i], 4 * c * c);
    conv(f2w[i], f2wbf[i], 4 * c * c);
  }

  // ---- layernorms (q_i and fused concat KV) ----
  for (int i = 0; i < 4; ++i)
    ln_rows_kernel<<<T, 256, 0, stream>>>(X[i], ln1g[i], ln1b[i], qbf[i],
                                          CSv[i], 1e-6f);
  ln_kv_kernel<<<T, 256, 0, stream>>>(X[0], X[1], X[2], X[3], lnkvg, lnkvb,
                                      kvbf, 1e-6f);

  // ---- KV path: kvp = (kv @ Wk^T) @ Wv^T, per head ----
  gemm(kvbf, wkbf, nullptr, t1bf, nullptr, nullptr, nullptr,
       512, 960, 960, (s64)512 * 960, (s64)960 * 960, 8,
       (s64)512 * 960, 0, Z, 1.f, GF_OUTB);
  gemm(t1bf, wvbf, nullptr, kvpbf, nullptr, nullptr, nullptr,
       512, 960, 960, (s64)512 * 960, (s64)960 * 960, 8,
       (s64)512 * 960, 0, Z, 1.f, GF_OUTB);
  {
    dim3 tg(960 / 32, 512 / 32, Z);
    transpose_bf16_kernel<<<tg, 256, 0, stream>>>(kvpbf, ktbf, 512, 960);
  }

  const float invs = 1.0f / sqrtf(960.f);
  s64 outoff = 0;
  for (int i = 0; i < 4; ++i) {
    const int c = CSv[i];
    // qp = qh @ Wq^T            [Z][512][c]
    gemm(qbf[i], wqbf[i], nullptr, qpbf, nullptr, nullptr, nullptr,
         512, c, c, (s64)512 * c, (s64)c * c, 8, (s64)512 * c, 0, Z, 1.f,
         GF_OUTB);
    // qp^T                       [Z][c][512]
    {
      dim3 tg((c + 31) / 32, 512 / 32, Z);
      transpose_bf16_kernel<<<tg, 256, 0, stream>>>(qpbf, qptbf, 512, c);
    }
    // scores = (qp^T)(K^T)^T * inv   [Z][c][960], K-dim = n = 512
    gemm(qptbf, ktbf, scoresf, nullptr, nullptr, nullptr, nullptr,
         c, 960, 512, (s64)c * 512, (s64)960 * 512, Z, (s64)c * 960, 0, Z,
         invs, GF_OUTF);
    // instance-norm stats + softmax -> prob (bf16)
    inorm_stats_kernel<<<Z, 256, 0, stream>>>(scoresf, stats, (s64)c * 960);
    inorm_softmax_kernel<<<Z * c, 256, 0, stream>>>(scoresf, stats, probbf, c);
    // o = V @ prob^T              [Z][512][c] == [T][c]
    gemm(kvpbf, probbf, nullptr, obf, nullptr, nullptr, nullptr,
         512, c, 960, (s64)512 * 960, (s64)c * 960, Z, (s64)512 * c, 0, Z,
         1.f, GF_OUTB);
    // attn = o @ Wo^T + q_i (residual on post-LN q)   [T][c] fp32
    gemm(obf, wobf[i], attnf, nullptr, nullptr, nullptr, qbf[i],
         T, c, c, 0, 0, 1, 0, 0, 1, 1.f, GF_RESB | GF_OUTF);
    // ln2(attn) -> bf16
    ln_rows_kernel<<<T, 256, 0, stream>>>(attnf, ln2g[i], ln2b[i], ln2bf, c,
                                          1e-6f);
    // h = gelu(ln2 @ fc1^T + b1)  [T][4c]
    gemm(ln2bf, f1wbf[i], nullptr, hbf, f1b[i], nullptr, nullptr,
         T, 4 * c, c, 0, 0, 1, 0, 0, 1, 1.f, GF_BIAS | GF_GELU | GF_OUTB);
    // out_i = h @ fc2^T + b2 + attn    -> d_out (fp32)
    float* outp = (float*)d_out + outoff;
    gemm(hbf, f2wbf[i], outp, nullptr, f2b[i], attnf, nullptr,
         T, c, 4 * c, 0, 0, 1, 0, 0, 1, 1.f, GF_BIAS | GF_RESF | GF_OUTF);
    outoff += (s64)T * c;
  }
}
